// Attention_38388417692233
// MI455X (gfx1250) — matrix operand order repaired and checked by reading, not compiled, not run on hardware
//
#include <hip/hip_runtime.h>
#include <hip/hip_bf16.h>

// ---------------------------------------------------------------------------
// MI455X (gfx1250) attention forward, wave32 + v_wmma_f32_16x16x32_bf16.
//   B=8, N=2048, C=320, H=5, D=64, scale = 1/8.
// Pipeline: qkv_gemm -> flash attention -> proj_gemm, intermediates bf16 in ws
// (42 MB, fits in 192MB L2 so inter-kernel traffic stays on-chip).
// Native __bf16 converts, base-2 online softmax (raw v_exp_f32), and ASYNC
// global->LDS staging (GLOBAL_LOAD_ASYNC_TO_LDS_B128 + s_wait_asynccnt) for K/V.
// ---------------------------------------------------------------------------

typedef __bf16 bf16_t;
typedef __bf16 v16bf __attribute__((ext_vector_type(16)));
typedef float  v8f   __attribute__((ext_vector_type(8)));
typedef int    v4i_t __attribute__((vector_size(16)));   // matches builtin's v4i

union FragB16 {
  v16bf  v;
  bf16_t u[16];
  uint4  q[2];
};

#define BATCH 8
#define SEQ   2048
#define CDIM  320
#define HEADS 5
#define HDIM  64
// softmax computed base-2: fold attn scale (1/8) * log2(e) into one multiplier
#define SCALE_LOG2E 0.18033688011112042f

#if __has_builtin(__builtin_amdgcn_exp2f)
#define EXP2(x) __builtin_amdgcn_exp2f(x)
#else
#define EXP2(x) exp2f(x)
#endif

#if __has_builtin(__builtin_amdgcn_rcpf)
#define RCP(x) __builtin_amdgcn_rcpf(x)
#else
#define RCP(x) (1.0f / (x))
#endif

#if defined(__gfx1250__) && __has_builtin(__builtin_amdgcn_global_load_async_to_lds_b128)
#define HAS_ASYNC_LDS 1
#define ASG __attribute__((address_space(1)))
#define ASL __attribute__((address_space(3)))
#else
#define HAS_ASYNC_LDS 0
#endif

// ---------------------------------------------------------------------------
// Kernel 1: QKV projection. Y[16384,960] = X[16384,320] * Wqkv^T (NT gemm).
// Block = 256 thr (8 waves), tile 128(M) x 64(N), K-step 32.
// Output scattered to Q/K/V bf16 buffers laid out [B,H,N,D] row-major.
// ---------------------------------------------------------------------------
__global__ __launch_bounds__(256) void qkv_gemm_k(
    const float* __restrict__ X, const float* __restrict__ W,
    bf16_t* __restrict__ Qb, bf16_t* __restrict__ Kb, bf16_t* __restrict__ Vb)
{
  __shared__ bf16_t Xs[128 * 40];   // 128 x 32 bf16, stride 40 (16B-aligned runs)
  __shared__ bf16_t Ws[64 * 40];

  const int tid  = threadIdx.x;
  const int lane = tid & 31;
  const int wave = tid >> 5;
  const int hl   = lane >> 4;    // which 16-lane half
  const int l16  = lane & 15;
  const int n0   = blockIdx.x * 64;    // output col base: 0..896
  const int m0   = blockIdx.y * 128;   // row base: 0..16256

  v8f acc[4] = {};

  for (int kc = 0; kc < CDIM; kc += 32) {
    __syncthreads();
    // Stage X chunk 128x32 f32 -> bf16 (1024 float4, 4 per thread)
    #pragma unroll
    for (int p = 0; p < 4; ++p) {
      int j = tid + p * 256;
      int r = j >> 3, c4 = j & 7;
      const float4 f = *(const float4*)(X + (size_t)(m0 + r) * CDIM + kc + c4 * 4);
      bf16_t* d = &Xs[r * 40 + c4 * 4];
      d[0] = (bf16_t)f.x; d[1] = (bf16_t)f.y; d[2] = (bf16_t)f.z; d[3] = (bf16_t)f.w;
    }
    // Stage W chunk 64x32 f32 -> bf16 (512 float4, 2 per thread)
    #pragma unroll
    for (int p = 0; p < 2; ++p) {
      int j = tid + p * 256;
      int r = j >> 3, c4 = j & 7;
      const float4 f = *(const float4*)(W + (size_t)(n0 + r) * CDIM + kc + c4 * 4);
      bf16_t* d = &Ws[r * 40 + c4 * 4];
      d[0] = (bf16_t)f.x; d[1] = (bf16_t)f.y; d[2] = (bf16_t)f.z; d[3] = (bf16_t)f.w;
    }
    __syncthreads();

    // A fragment: 16x32, lane half picks K = {hl*8..+7, hl*8+16..+23}
    FragB16 a;
    {
      const bf16_t* src = &Xs[(wave * 16 + l16) * 40 + hl * 8];
      a.q[0] = *(const uint4*)(src);
      a.q[1] = *(const uint4*)(src + 16);
    }
    // 4 B fragments (32x16 slices of W^T) + WMMAs
    #pragma unroll
    for (int t = 0; t < 4; ++t) {
      FragB16 bf;
      const bf16_t* src = &Ws[(t * 16 + l16) * 40 + hl * 8];
      bf.q[0] = *(const uint4*)(src);
      bf.q[1] = *(const uint4*)(src + 16);
      acc[t] = __builtin_amdgcn_wmma_f32_16x16x32_bf16(
          false, a.v, false, bf.v, (short)0, acc[t], false, false);
    }
  }

  // Scatter to Q/K/V [B,H,N,D] bf16. A 64-wide col tile lies in one (part, head).
  const int part = n0 / CDIM;
  const int head = (n0 % CDIM) / HDIM;
  bf16_t* dst = (part == 0) ? Qb : (part == 1) ? Kb : Vb;
  #pragma unroll
  for (int t = 0; t < 4; ++t) {
    const int d = t * 16 + l16;
    #pragma unroll
    for (int i = 0; i < 8; ++i) {
      int r = hl * 8 + i;
      int m = m0 + wave * 16 + r;
      int b = m >> 11, n = m & (SEQ - 1);
      dst[(((size_t)(b * HEADS + head) * SEQ + n) * HDIM) + d] = (bf16_t)acc[t][i];
    }
  }
}

// ---------------------------------------------------------------------------
// Kernel 2: flash attention. grid = (N/128, B*H); 8 waves x 16 query rows.
// Key loop chunks of 64: S = Q K^T (8 wmma), online softmax, O += P V (8 wmma).
// K/V chunks staged to LDS with GLOBAL_LOAD_ASYNC_TO_LDS_B128 when available.
// ---------------------------------------------------------------------------
__global__ __launch_bounds__(256) void attn_fwd_k(
    const bf16_t* __restrict__ Qb, const bf16_t* __restrict__ Kb,
    const bf16_t* __restrict__ Vb, bf16_t* __restrict__ AO)
{
  __shared__ bf16_t Kl[64 * 72];        // 64 keys x 64 d, padded stride 72
  __shared__ bf16_t Vl[64 * 72];
  __shared__ bf16_t Pl[8][16 * 72];     // per-wave P tile 16x64

  const int tid  = threadIdx.x;
  const int lane = tid & 31;
  const int wave = tid >> 5;
  const int hl   = lane >> 4;
  const int l16  = lane & 15;
  const int bh   = blockIdx.y;                 // 0..39  (b*H + h)
  const int qb0  = blockIdx.x * 128 + wave * 16;
  const size_t base = (size_t)bh * SEQ * HDIM;

  // Q fragments held in registers for whole key loop (16 rows x 64 d = 2 frags)
  FragB16 qf[2];
  {
    const bf16_t* src = Qb + base + (size_t)(qb0 + l16) * HDIM;
    #pragma unroll
    for (int s = 0; s < 2; ++s) {
      const bf16_t* p = src + s * 32 + hl * 8;
      qf[s].q[0] = *(const uint4*)p;
      qf[s].q[1] = *(const uint4*)(p + 16);
    }
  }

  v8f o[4] = {};
  float mrow[8], lrow[8];
  #pragma unroll
  for (int i = 0; i < 8; ++i) { mrow[i] = -1e30f; lrow[i] = 0.f; }

  for (int kc = 0; kc < SEQ; kc += 64) {
    __syncthreads();                           // protect Kl/Vl from prior reads
    // Cooperative staging of K,V chunk (already bf16): 512x16B each, 2/thread
    #pragma unroll
    for (int p = 0; p < 2; ++p) {
      int j = tid + p * 256;
      int r = j >> 3, c = j & 7;
      const uint4* kg = (const uint4*)(Kb + base + (size_t)(kc + r) * HDIM) + c;
      const uint4* vg = (const uint4*)(Vb + base + (size_t)(kc + r) * HDIM) + c;
#if HAS_ASYNC_LDS
      __builtin_amdgcn_global_load_async_to_lds_b128(
          (ASG v4i_t*)kg, (ASL v4i_t*)&Kl[r * 72 + c * 8], 0, 0);
      __builtin_amdgcn_global_load_async_to_lds_b128(
          (ASG v4i_t*)vg, (ASL v4i_t*)&Vl[r * 72 + c * 8], 0, 0);
#else
      *(uint4*)&Kl[r * 72 + c * 8] = *kg;
      *(uint4*)&Vl[r * 72 + c * 8] = *vg;
#endif
    }
#if HAS_ASYNC_LDS
#if __has_builtin(__builtin_amdgcn_s_wait_asynccnt)
    __builtin_amdgcn_s_wait_asynccnt(0);
#else
    asm volatile("s_wait_asynccnt 0" ::: "memory");
#endif
#endif
    __syncthreads();

    // S = Q * K^T  (4 key tiles x 2 k-steps)
    v8f s[4] = {};
    #pragma unroll
    for (int t = 0; t < 4; ++t) {
      #pragma unroll
      for (int ss = 0; ss < 2; ++ss) {
        FragB16 kf;   // B frag: col = key (t*16+l16), K = d = ss*32 + {hl*8..+7, hl*8+16..+23}
        const bf16_t* p = &Kl[(t * 16 + l16) * 72 + ss * 32 + hl * 8];
        kf.q[0] = *(const uint4*)p;
        kf.q[1] = *(const uint4*)(p + 16);
        s[t] = __builtin_amdgcn_wmma_f32_16x16x32_bf16(
            false, qf[ss].v, false, kf.v, (short)0, s[t], false, false);
      }
    }

    // Online softmax in base-2 domain. In C-layout, row (hl*8+i) lives in
    // VGPR i across a 16-lane half -> shfl_xor 8/4/2/1 reduces that row.
    #pragma unroll
    for (int i = 0; i < 8; ++i) {
      float mx = -1e30f;
      #pragma unroll
      for (int t = 0; t < 4; ++t) {
        float v0 = s[t][i] * SCALE_LOG2E;
        s[t][i] = v0;
        mx = fmaxf(mx, v0);
      }
      #pragma unroll
      for (int off = 8; off >= 1; off >>= 1) mx = fmaxf(mx, __shfl_xor(mx, off, 32));
      float mnew  = fmaxf(mrow[i], mx);
      float alpha = EXP2(mrow[i] - mnew);
      mrow[i] = mnew;
      float rs = 0.f;
      #pragma unroll
      for (int t = 0; t < 4; ++t) {
        float pe = EXP2(s[t][i] - mnew);
        s[t][i] = pe;
        rs += pe;
      }
      #pragma unroll
      for (int off = 8; off >= 1; off >>= 1) rs += __shfl_xor(rs, off, 32);
      lrow[i] = lrow[i] * alpha + rs;
      #pragma unroll
      for (int t = 0; t < 4; ++t) o[t][i] *= alpha;
      const int r = hl * 8 + i;
      #pragma unroll
      for (int t = 0; t < 4; ++t) Pl[wave][r * 72 + t * 16 + l16] = (bf16_t)s[t][i];
    }
    __syncthreads();

    // O += P * V  (2 key-steps x 4 d tiles)
    #pragma unroll
    for (int ss = 0; ss < 2; ++ss) {
      FragB16 pf;   // A frag of P: row = l16, K = key
      const bf16_t* pp = &Pl[wave][l16 * 72 + ss * 32 + hl * 8];
      pf.q[0] = *(const uint4*)pp;
      pf.q[1] = *(const uint4*)(pp + 16);
      #pragma unroll
      for (int t = 0; t < 4; ++t) {
        FragB16 vf;  // B frag: col = d (t*16+l16), K = key = ss*32 + hl*8 + (e&7) + (e>>3)*16
        #pragma unroll
        for (int e = 0; e < 16; ++e)
          vf.u[e] = Vl[(ss * 32 + hl * 8 + (e & 7) + (e >> 3) * 16) * 72 + t * 16 + l16];
        o[t] = __builtin_amdgcn_wmma_f32_16x16x32_bf16(
            false, pf.v, false, vf.v, (short)0, o[t], false, false);
      }
    }
  }

  // Normalize and store to AO laid out [B, N, H*D] (= rows for proj gemm)
  const int b = bh / HEADS, h = bh % HEADS;
  #pragma unroll
  for (int i = 0; i < 8; ++i) {
    float inv = RCP(lrow[i]);
    int r = hl * 8 + i;
    int n = qb0 + r;
    #pragma unroll
    for (int t = 0; t < 4; ++t) {
      int d = t * 16 + l16;
      AO[((size_t)(b * SEQ + n)) * CDIM + h * HDIM + d] = (bf16_t)(o[t][i] * inv);
    }
  }
}

// ---------------------------------------------------------------------------
// Kernel 3: out[16384,320] = AO * Wproj^T + bias  (f32 output)
// ---------------------------------------------------------------------------
__global__ __launch_bounds__(256) void proj_gemm_k(
    const bf16_t* __restrict__ AO, const float* __restrict__ W,
    const float* __restrict__ bias, float* __restrict__ out)
{
  __shared__ bf16_t As[128 * 40];
  __shared__ bf16_t Ws[64 * 40];

  const int tid  = threadIdx.x;
  const int lane = tid & 31;
  const int wave = tid >> 5;
  const int hl   = lane >> 4;
  const int l16  = lane & 15;
  const int n0   = blockIdx.x * 64;    // 0..256
  const int m0   = blockIdx.y * 128;

  v8f acc[4] = {};

  for (int kc = 0; kc < CDIM; kc += 32) {
    __syncthreads();
    // A chunk: 128 rows x 32 bf16 = 512 uint4, 2 per thread (already bf16)
    #pragma unroll
    for (int p = 0; p < 2; ++p) {
      int j = tid + p * 256;
      int r = j >> 2, c = j & 3;
      *(uint4*)&As[r * 40 + c * 8] =
          *((const uint4*)(AO + (size_t)(m0 + r) * CDIM + kc) + c);
    }
    // W chunk 64x32 f32 -> bf16
    #pragma unroll
    for (int p = 0; p < 2; ++p) {
      int j = tid + p * 256;
      int r = j >> 3, c4 = j & 7;
      const float4 f = *(const float4*)(W + (size_t)(n0 + r) * CDIM + kc + c4 * 4);
      bf16_t* d = &Ws[r * 40 + c4 * 4];
      d[0] = (bf16_t)f.x; d[1] = (bf16_t)f.y; d[2] = (bf16_t)f.z; d[3] = (bf16_t)f.w;
    }
    __syncthreads();

    FragB16 a;
    {
      const bf16_t* src = &As[(wave * 16 + l16) * 40 + hl * 8];
      a.q[0] = *(const uint4*)(src);
      a.q[1] = *(const uint4*)(src + 16);
    }
    #pragma unroll
    for (int t = 0; t < 4; ++t) {
      FragB16 bf;
      const bf16_t* src = &Ws[(t * 16 + l16) * 40 + hl * 8];
      bf.q[0] = *(const uint4*)(src);
      bf.q[1] = *(const uint4*)(src + 16);
      acc[t] = __builtin_amdgcn_wmma_f32_16x16x32_bf16(
          false, a.v, false, bf.v, (short)0, acc[t], false, false);
    }
  }

  #pragma unroll
  for (int t = 0; t < 4; ++t) {
    int col = n0 + t * 16 + l16;
    float bv = bias[col];
    #pragma unroll
    for (int i = 0; i < 8; ++i) {
      int r = hl * 8 + i;
      int m = m0 + wave * 16 + r;
      out[(size_t)m * CDIM + col] = acc[t][i] + bv;
    }
  }
}

// ---------------------------------------------------------------------------
// Workspace layout (bf16 elements): Q, K, V, AO each 5,242,880 elems -> 42 MB.
// ---------------------------------------------------------------------------
extern "C" void kernel_launch(void* const* d_in, const int* in_sizes, int n_in,
                              void* d_out, int out_size, void* d_ws, size_t ws_size,
                              hipStream_t stream) {
  const float* x      = (const float*)d_in[0];
  const float* w_qkv  = (const float*)d_in[1];
  const float* w_proj = (const float*)d_in[2];
  const float* b_proj = (const float*)d_in[3];
  float* out = (float*)d_out;

  const size_t QKV_ELEMS = (size_t)BATCH * HEADS * SEQ * HDIM;  // 5,242,880
  bf16_t* Qb = (bf16_t*)d_ws;
  bf16_t* Kb = Qb + QKV_ELEMS;
  bf16_t* Vb = Kb + QKV_ELEMS;
  bf16_t* AO = Vb + QKV_ELEMS;

  // 1) QKV projection: grid (960/64, 16384/128)
  qkv_gemm_k<<<dim3(15, 128), 256, 0, stream>>>(x, w_qkv, Qb, Kb, Vb);
  // 2) Flash attention: grid (2048/128, B*H)
  attn_fwd_k<<<dim3(16, 40), 256, 0, stream>>>(Qb, Kb, Vb, AO);
  // 3) Output projection + bias: grid (320/64, 16384/128)
  proj_gemm_k<<<dim3(5, 128), 256, 0, stream>>>(AO, w_proj, b_proj, out);
}
